// BorderBranch_21234318311586
// MI455X (gfx1250) — matrix operand order repaired and checked by reading, not compiled, not run on hardware
//
#include <hip/hip_runtime.h>
#include <hip/hip_bf16.h>
#include <stdint.h>

typedef __attribute__((ext_vector_type(16))) __bf16 v16bf;
typedef __attribute__((ext_vector_type(8)))  float  v8f;

union FragBF {
  v16bf v;
  unsigned int u32[8];
  unsigned short u16[16];
};

__device__ __forceinline__ unsigned short f2bf(float f) {
  unsigned int u = __float_as_uint(f);
  unsigned int r = (u + 0x7FFFu + ((u >> 16) & 1u)) >> 16;  // RNE
  return (unsigned short)r;
}
__device__ __forceinline__ unsigned int f2bf2(float lo, float hi) {
  return (unsigned int)f2bf(lo) | ((unsigned int)f2bf(hi) << 16);
}

// ---------------------------------------------------------------------------
// f32 -> bf16 pack (4 elems / thread)
// ---------------------------------------------------------------------------
__global__ void cast_f32_bf16(const float* __restrict__ src,
                              unsigned short* __restrict__ dst, int n4) {
  int i = blockIdx.x * blockDim.x + threadIdx.x;
  if (i < n4) {
    float4 f = ((const float4*)src)[i];
    uint2 o;
    o.x = f2bf2(f.x, f.y);
    o.y = f2bf2(f.z, f.w);
    ((uint2*)dst)[i] = o;
  }
}

// ---------------------------------------------------------------------------
// bf16 GEMM via V_WMMA_F32_16X16X32_BF16, double-buffered LDS pipeline.
//   C[img][m][hw] = sum_k A[m][k] * B[img][k][hw]   (hw dim = 4096)
// Block: 256 thr = 8 waves (4M x 2N). Block tile 64M x 128N, K step 32.
// A tiles are staged with GLOBAL_LOAD_ASYNC_TO_LDS_B128 (ASYNCcnt-tracked,
// VGPR-free); next B tile is prefetched into registers during WMMA and
// scatter-transposed into the ping-pong LDS buffer afterwards.
// If bias != nullptr: C = relu(C + bias[m]).
// ---------------------------------------------------------------------------
#define BM 64
#define BN 128
#define BK 32
#define LP 40   // LDS pitch in ushorts: 80B rows -> 16B aligned, conflict-free

template <int K>
__global__ __launch_bounds__(256) void gemm_bf16_wmma(
    const unsigned short* __restrict__ A,   // [M][K] bf16 weights
    const unsigned short* __restrict__ Bb,  // [Nimg][K][4096] bf16
    float* __restrict__ C,                  // [Nimg][M][4096] f32
    const float* __restrict__ bias,         // optional (+relu)
    int M) {
  __shared__ unsigned short ldsA[2][BM][LP];
  __shared__ unsigned short ldsB[2][BN][LP];   // transposed: [n][k]

  const int tid   = threadIdx.x;
  const int lane  = tid & 31;
  const int wid   = tid >> 5;
  const int warpM = wid & 3;
  const int warpN = wid >> 2;
  const int tileN = blockIdx.x * BN;
  const int tileM = blockIdx.y * BM;
  const int img   = blockIdx.z;

  const unsigned short* Bsrc = Bb + (size_t)img * K * 4096;
  float* Cdst = C + (size_t)img * M * 4096;

  const v8f vzero = {0.f, 0.f, 0.f, 0.f, 0.f, 0.f, 0.f, 0.f};
  v8f acc[4];
#pragma unroll
  for (int i = 0; i < 4; ++i) acc[i] = vzero;

  const int ml   = lane & 15;   // row/col within 16
  const int half = lane >> 4;   // K-half selector

  // per-thread staging coordinates (constant across K steps)
  const int arow = tid >> 2;            // A: 4 threads per 64-row, 16B each
  const int ack  = (tid & 3) * 8;
  const unsigned ldsA_dst[2] = {
      (unsigned)(uintptr_t)&ldsA[0][arow][ack],
      (unsigned)(uintptr_t)&ldsA[1][arow][ack]};
  const int bkk [2] = { tid >> 4,         (tid + 256) >> 4 };        // 0..31
  const int bnch[2] = { (tid & 15) * 8,   ((tid + 256) & 15) * 8 };  // 0..120

  constexpr int NS = K / BK;

  // ---- prologue: stage K-step 0 into buffer 0
  {
    const unsigned short* gA = A + (size_t)(tileM + arow) * K + ack;
    asm volatile("global_load_async_to_lds_b128 %0, %1, off"
                 :: "v"(ldsA_dst[0]), "v"(gA) : "memory");
#pragma unroll
    for (int c2 = 0; c2 < 2; ++c2) {
      uint4 q = *(const uint4*)(Bsrc + (size_t)bkk[c2] * 4096 + tileN + bnch[c2]);
      unsigned int w[4] = {q.x, q.y, q.z, q.w};
#pragma unroll
      for (int j = 0; j < 4; ++j) {
        ldsB[0][bnch[c2] + 2 * j    ][bkk[c2]] = (unsigned short)(w[j] & 0xFFFFu);
        ldsB[0][bnch[c2] + 2 * j + 1][bkk[c2]] = (unsigned short)(w[j] >> 16);
      }
    }
  }

#pragma unroll 2
  for (int ks = 0; ks < NS; ++ks) {
    const int buf = ks & 1;
    // own async A-transfer (into buf) done, then block-wide: buf fully staged
    // and all readers of buf^1 finished.
    asm volatile("s_wait_asynccnt 0x0" ::: "memory");
    __syncthreads();

    // ---- issue next tile: async A -> LDS(buf^1); B tile -> registers
    uint4 qn[2];
    const bool more = (ks + 1 < NS);
    if (more) {
      const int k0n = (ks + 1) * BK;
      const unsigned short* gA = A + (size_t)(tileM + arow) * K + k0n + ack;
      asm volatile("global_load_async_to_lds_b128 %0, %1, off"
                   :: "v"(ldsA_dst[buf ^ 1]), "v"(gA) : "memory");
#pragma unroll
      for (int c2 = 0; c2 < 2; ++c2)
        qn[c2] = *(const uint4*)(Bsrc + (size_t)(k0n + bkk[c2]) * 4096 +
                                 tileN + bnch[c2]);
    }

    // ---- compute on current buffer
    FragBF a;
    {
      const int ar = warpM * 16 + ml;
#pragma unroll
      for (int j = 0; j < 4; ++j)
        a.u32[j] = *(const unsigned int*)&ldsA[buf][ar][half * 8 + 2 * j];
#pragma unroll
      for (int j = 0; j < 4; ++j)
        a.u32[4 + j] = *(const unsigned int*)&ldsA[buf][ar][16 + half * 8 + 2 * j];
    }
#pragma unroll
    for (int nt = 0; nt < 4; ++nt) {
      FragBF b;
      const int brow = warpN * 64 + nt * 16 + ml;
#pragma unroll
      for (int j = 0; j < 8; ++j)
        b.u32[j] = *(const unsigned int*)&ldsB[buf][brow][(j >> 2) * 16 + half * 8 + 2 * (j & 3)];
      acc[nt] = __builtin_amdgcn_wmma_f32_16x16x32_bf16(
          false, a.v, false, b.v, (short)0, acc[nt], false, false);
    }

    // ---- scatter-transpose prefetched B registers into LDS(buf^1)
    if (more) {
#pragma unroll
      for (int c2 = 0; c2 < 2; ++c2) {
        unsigned int w[4] = {qn[c2].x, qn[c2].y, qn[c2].z, qn[c2].w};
#pragma unroll
        for (int j = 0; j < 4; ++j) {
          ldsB[buf ^ 1][bnch[c2] + 2 * j    ][bkk[c2]] = (unsigned short)(w[j] & 0xFFFFu);
          ldsB[buf ^ 1][bnch[c2] + 2 * j + 1][bkk[c2]] = (unsigned short)(w[j] >> 16);
        }
      }
    }
  }

  // --- store (ISA 32-bit C/D 16x16 layout: VGPR r, lane l -> M=r+8*(l>>4), N=l&15)
#pragma unroll
  for (int nt = 0; nt < 4; ++nt) {
    const int col = tileN + warpN * 64 + nt * 16 + ml;
#pragma unroll
    for (int r = 0; r < 8; ++r) {
      int mrow = tileM + warpM * 16 + r + 8 * half;
      float v = acc[nt][r];
      if (bias) v = fmaxf(v + bias[mrow], 0.0f);
      Cdst[(size_t)mrow * 4096 + col] = v;
    }
  }
}

// ---------------------------------------------------------------------------
// Instance norm + ReLU, one block per (n, channel) over 4096 spatial elems.
// c <  64 : fm_short -> write bf16 into catbf rows 256+c
// c >= 64 : border-align source -> write f32 back in place (conv12)
// ---------------------------------------------------------------------------
__global__ __launch_bounds__(256) void inorm_relu_kernel(
    float* __restrict__ conv12,          // [N][320][4096]
    unsigned short* __restrict__ catbf)  // [N][320][4096] bf16
{
  const int c = blockIdx.x % 320;
  const int n = blockIdx.x / 320;
  const int t = threadIdx.x;
  float* row = conv12 + ((size_t)n * 320 + c) * 4096;

  float4 x[4];
  float s = 0.f, s2 = 0.f;
#pragma unroll
  for (int j = 0; j < 4; ++j) {
    x[j] = ((const float4*)row)[t + 256 * j];
    s  += x[j].x + x[j].y + x[j].z + x[j].w;
    s2 += x[j].x * x[j].x + x[j].y * x[j].y + x[j].z * x[j].z + x[j].w * x[j].w;
  }
  __shared__ float rs[256], rs2[256];
  rs[t] = s; rs2[t] = s2;
  __syncthreads();
  for (int off = 128; off > 0; off >>= 1) {
    if (t < off) { rs[t] += rs[t + off]; rs2[t] += rs2[t + off]; }
    __syncthreads();
  }
  const float mean = rs[0] * (1.0f / 4096.0f);
  const float var  = rs2[0] * (1.0f / 4096.0f) - mean * mean;
  const float rstd = rsqrtf(var + 1e-5f);

  if (c < 64) {
    unsigned short* drow = catbf + ((size_t)n * 320 + 256 + c) * 4096;
#pragma unroll
    for (int j = 0; j < 4; ++j) {
      float a0 = fmaxf((x[j].x - mean) * rstd, 0.f);
      float a1 = fmaxf((x[j].y - mean) * rstd, 0.f);
      float a2 = fmaxf((x[j].z - mean) * rstd, 0.f);
      float a3 = fmaxf((x[j].w - mean) * rstd, 0.f);
      uint2 o; o.x = f2bf2(a0, a1); o.y = f2bf2(a2, a3);
      ((uint2*)drow)[t + 256 * j] = o;
    }
  } else {
#pragma unroll
    for (int j = 0; j < 4; ++j) {
      float4 y;
      y.x = fmaxf((x[j].x - mean) * rstd, 0.f);
      y.y = fmaxf((x[j].y - mean) * rstd, 0.f);
      y.z = fmaxf((x[j].z - mean) * rstd, 0.f);
      y.w = fmaxf((x[j].w - mean) * rstd, 0.f);
      ((float4*)row)[t + 256 * j] = y;
    }
  }
}

// ---------------------------------------------------------------------------
// Border align: one wave per (n, box, border-group); lane covers channels
// c=lane and c=lane+32 of the 64-channel group. Max over 11 bilinear samples.
// Output bf16 -> catbf rows g*64+c, column = box index (== hw).
// ---------------------------------------------------------------------------
__global__ __launch_bounds__(256) void border_align_kernel(
    const float* __restrict__ fsamp,     // conv12: [N][320][4096], groups at 64+
    const float* __restrict__ boxes,     // [N][4096][4]
    unsigned short* __restrict__ catbf)  // [N][320][4096]
{
  const int gw   = blockIdx.x * 8 + (threadIdx.x >> 5);
  const int lane = threadIdx.x & 31;
  const int g = gw & 3;
  const int b = (gw >> 2) & 4095;
  const int n = gw >> 14;

  const float4 bx = ((const float4*)boxes)[(size_t)n * 4096 + b];
  const float x1 = bx.x, y1 = bx.y, x2 = bx.z, y2 = bx.w;

  const float* f0 = fsamp + ((size_t)n * 320 + 64 + g * 64 + lane) * 4096;
  const float* f1 = f0 + 32 * 4096;

  float m0 = -1e30f, m1 = -1e30f;
#pragma unroll
  for (int p = 0; p <= 10; ++p) {
    const float tt = (float)p * 0.1f;
    float x, y;
    if (g == 0)      { x = x1 + (x2 - x1) * tt; y = y1; }
    else if (g == 1) { x = x1;                  y = y1 + (y2 - y1) * tt; }
    else if (g == 2) { x = x1 + (x2 - x1) * tt; y = y2; }
    else             { x = x2;                  y = y1 + (y2 - y1) * tt; }
    const bool invalid = (y < -1.f) | (y > 64.f) | (x < -1.f) | (x > 64.f);
    float yc = fmaxf(y, 0.f), xc = fmaxf(x, 0.f);
    int yl = min(max((int)floorf(yc), 0), 63);
    int xl = min(max((int)floorf(xc), 0), 63);
    int yh = min(yl + 1, 63);
    int xh = min(xl + 1, 63);
    float ly = yc - (float)yl, lx = xc - (float)xl;
    float hy = 1.f - ly, hx = 1.f - lx;
    float w00 = hy * hx, w01 = hy * lx, w10 = ly * hx, w11 = ly * lx;
    int i00 = yl * 64 + xl, i01 = yl * 64 + xh;
    int i10 = yh * 64 + xl, i11 = yh * 64 + xh;
    float v0 = w00 * f0[i00] + w01 * f0[i01] + w10 * f0[i10] + w11 * f0[i11];
    float v1 = w00 * f1[i00] + w01 * f1[i01] + w10 * f1[i10] + w11 * f1[i11];
    if (invalid) { v0 = 0.f; v1 = 0.f; }
    m0 = fmaxf(m0, v0);
    m1 = fmaxf(m1, v1);
  }
  unsigned short* dst = catbf + ((size_t)n * 320 + g * 64 + lane) * 4096 + b;
  dst[0]          = f2bf(m0);
  dst[32 * 4096]  = f2bf(m1);
}

// ---------------------------------------------------------------------------
extern "C" void kernel_launch(void* const* d_in, const int* in_sizes, int n_in,
                              void* d_out, int out_size, void* d_ws, size_t ws_size,
                              hipStream_t stream) {
  (void)in_sizes; (void)n_in; (void)out_size; (void)ws_size;
  const float* feature = (const float*)d_in[0];
  const float* boxes   = (const float*)d_in[1];
  // d_in[2] = wh : unused by reference
  const float* W1 = (const float*)d_in[3];
  // d_in[4] = b1 : zero-effect before instance norm
  const float* W2 = (const float*)d_in[5];
  // d_in[6] = b2 : zero-effect before instance norm
  const float* W3 = (const float*)d_in[7];
  const float* b3 = (const float*)d_in[8];
  float* out = (float*)d_out;

  // workspace layout
  unsigned short* Fbf   = (unsigned short*)d_ws;           // 4*256*4096 bf16
  unsigned short* Wbf12 = Fbf + (size_t)4 * 256 * 4096;    // 320*256 bf16
  unsigned short* Wbf3  = Wbf12 + 320 * 256;               // 256*320 bf16
  unsigned short* catbf = Wbf3 + 256 * 320;                // 4*320*4096 bf16
  float* conv12 = (float*)(catbf + (size_t)4 * 320 * 4096);// 4*320*4096 f32

  // 1) bf16 conversions
  {
    int n4 = 4 * 256 * 4096 / 4;
    cast_f32_bf16<<<(n4 + 255) / 256, 256, 0, stream>>>(feature, Fbf, n4);
    n4 = 64 * 256 / 4;
    cast_f32_bf16<<<(n4 + 255) / 256, 256, 0, stream>>>(W1, Wbf12, n4);
    n4 = 256 * 256 / 4;
    cast_f32_bf16<<<(n4 + 255) / 256, 256, 0, stream>>>(W2, Wbf12 + 64 * 256, n4);
    n4 = 256 * 320 / 4;
    cast_f32_bf16<<<(n4 + 255) / 256, 256, 0, stream>>>(W3, Wbf3, n4);
  }

  // 2) fused conv1+conv2 GEMM: [320x256] x [256x4096] per image
  {
    dim3 grid(4096 / BN, 320 / BM, 4);
    gemm_bf16_wmma<256><<<grid, 256, 0, stream>>>(Wbf12, Fbf, conv12, nullptr, 320);
  }

  // 3) instance norm + relu (fm_short -> catbf bf16; f -> in-place f32)
  inorm_relu_kernel<<<4 * 320, 256, 0, stream>>>(conv12, catbf);

  // 4) border align -> catbf rows 0..255 (bf16)
  border_align_kernel<<<4 * 4096 * 4 / 8, 256, 0, stream>>>(conv12, boxes, catbf);

  // 5) conv3 GEMM + bias + relu -> d_out: [256x320] x [320x4096] per image
  {
    dim3 grid(4096 / BN, 256 / BM, 4);
    gemm_bf16_wmma<320><<<grid, 256, 0, stream>>>(Wbf3, catbf, out, b3, 256);
  }
}
